// ESA_blcok_37117107372567
// MI455X (gfx1250) — matrix operand order repaired and checked by reading, compiled, not run on hardware
//
#include <hip/hip_runtime.h>
#include <hip/hip_fp16.h>

typedef __attribute__((ext_vector_type(16))) _Float16 v16h;
typedef __attribute__((ext_vector_type(2)))  _Float16 h2;
typedef __attribute__((ext_vector_type(8)))  float    v8f;

__device__ __forceinline__ v8f wmma_f16(v16h a, v16h b, v8f c) {
  // D = A(16x32 f16) x B(32x16 f16) + C(16x16 f32)
  return __builtin_amdgcn_wmma_f32_16x16x32_f16(false, a, false, b, (short)0, c,
                                                false, false);
}

enum { EPI_NONE = 0, EPI_BNRELU = 1, EPI_GELU = 2, EPI_RES_STRIDED = 3, EPI_RES_ROW_TRANS = 4 };

// Tiled GEMM, all-f16 operands, f32 accumulate:
//   Out[b, t, o] = epi( sum_c X[b, t, c] * W[o, c] )
// X element (t,c) at Xb[t*ldx_t + c*ldx_c] (f16); W row-major f16 (N x K), ldw=K.
// Block: 256 threads = 8 waves in 4(M) x 2(N); block tile 64x64; each wave
// computes 16x32 (A fragment reused across two WMMAs); K-step 32; double-
// buffered LDS with reg-staged global loads (1 barrier / K-step).
template <int EPI, typename OT>
__global__ __launch_bounds__(256) void gemm_kernel(
    const _Float16* __restrict__ X, long long x_bstride, int ldx_t, int ldx_c,
    const _Float16* __restrict__ W, int ldw,
    OT* __restrict__ Out, long long out_bstride, int ld_out,
    int M, int N, int K,
    const float* __restrict__ p0, const float* __restrict__ p1,
    const float* __restrict__ p2, const float* __restrict__ p3,
    const float* __restrict__ p4,
    long long aux_bstride, int aux_stride) {
  const int tid  = threadIdx.x;
  const int lane = tid & 31;
  const int wv   = tid >> 5;   // 0..7
  const int wm   = wv >> 1;    // 0..3 (M)
  const int wn   = wv & 1;     // 0..1 (N)
  const int lm   = lane & 15;
  const int lh   = lane >> 4;
  const int b    = blockIdx.z;
  const int m0   = blockIdx.y * 64;
  const int n0   = blockIdx.x * 64;

  const _Float16* Xb = X + (long long)b * x_bstride;

  __shared__ _Float16 sa[2][64][40];   // [buf][row][k]   (k-contiguous)
  __shared__ _Float16 sbT[2][64][40];  // [buf][n][k]     (k-contiguous)

  v8f acc0 = {}, acc1 = {};
  _Float16 areg[8], breg[8];

  const int nk = K >> 5;

  // prologue: stage tile 0
  #pragma unroll
  for (int j = 0; j < 8; ++j) {
    int e = tid + j * 256;
    int r = e >> 5, k = e & 31;
    int t = m0 + r;
    areg[j] = (t < M) ? Xb[(long long)t * ldx_t + (long long)k * ldx_c] : (_Float16)0;
    breg[j] = W[(long long)(n0 + r) * ldw + k];
  }
  #pragma unroll
  for (int j = 0; j < 8; ++j) {
    int e = tid + j * 256;
    sa[0][e >> 5][e & 31]  = areg[j];
    sbT[0][e >> 5][e & 31] = breg[j];
  }
  __syncthreads();

  for (int it = 0; it < nk; ++it) {
    const int cur = it & 1;
    const bool more = (it + 1) < nk;
    if (more) {
      const int k0 = (it + 1) << 5;
      #pragma unroll
      for (int j = 0; j < 8; ++j) {
        int e = tid + j * 256;
        int r = e >> 5, k = e & 31;
        int t = m0 + r;
        areg[j] = (t < M) ? Xb[(long long)t * ldx_t + (long long)(k0 + k) * ldx_c]
                          : (_Float16)0;
        breg[j] = W[(long long)(n0 + r) * ldw + (k0 + k)];
      }
    }

    // A fragment (ISA 7.12.2): lane row = wm*16 + lm; K-half by lane/16
    v16h af, bf0, bf1;
    const int arow = wm * 16 + lm;
    #pragma unroll
    for (int g = 0; g < 8; ++g) {
      int kk = (g < 4) ? (lh * 8 + g * 2) : (16 + lh * 8 + (g - 4) * 2);
      h2 p = *(const h2*)&sa[cur][arow][kk];
      af[2 * g] = p.x; af[2 * g + 1] = p.y;
    }
    // B fragments: lane col = wn*32 + {0,16} + lm; K-half by lane/16
    const int bc0 = wn * 32 + lm;
    #pragma unroll
    for (int g = 0; g < 8; ++g) {
      int kk = (g < 4) ? (lh * 8 + g * 2) : (16 + lh * 8 + (g - 4) * 2);
      h2 p = *(const h2*)&sbT[cur][bc0][kk];
      bf0[2 * g] = p.x; bf0[2 * g + 1] = p.y;
      h2 q = *(const h2*)&sbT[cur][bc0 + 16][kk];
      bf1[2 * g] = q.x; bf1[2 * g + 1] = q.y;
    }
    acc0 = wmma_f16(af, bf0, acc0);
    acc1 = wmma_f16(af, bf1, acc1);

    if (more) {
      const int nb = cur ^ 1;
      #pragma unroll
      for (int j = 0; j < 8; ++j) {
        int e = tid + j * 256;
        sa[nb][e >> 5][e & 31]  = areg[j];
        sbT[nb][e >> 5][e & 31] = breg[j];
      }
    }
    __syncthreads();
  }

  // epilogue; C layout: M = g + 8*(lane/16), N = lane%16
  #pragma unroll
  for (int s = 0; s < 2; ++s) {
    const int o = n0 + wn * 32 + s * 16 + lm;
    #pragma unroll
    for (int g = 0; g < 8; ++g) {
      int t = m0 + wm * 16 + 8 * lh + g;
      if (t >= M) continue;
      float v = s ? acc1[g] : acc0[g];
      if (EPI == EPI_NONE) {
        if (p0) v += p0[o];
        Out[(long long)b * out_bstride + (long long)t * ld_out + o] = (OT)v;
      } else if (EPI == EPI_BNRELU) {
        v += p0[o];
        float sc = p1[o] * rsqrtf(p4[o] + 1e-5f);
        v = (v - p3[o]) * sc + p2[o];
        Out[(long long)b * out_bstride + (long long)t * ld_out + o] = (OT)(v > 0.f ? v : 0.f);
      } else if (EPI == EPI_GELU) {
        v += p0[o];
        v = 0.5f * v * (1.f + erff(v * 0.70710678118654752f));
        Out[(long long)b * out_bstride + (long long)t * ld_out + o] = (OT)v;
      } else if (EPI == EPI_RES_STRIDED) {
        v += p0[o] + p1[(long long)b * aux_bstride + (long long)o * aux_stride + t];
        Out[(long long)b * out_bstride + (long long)t * ld_out + o] = (OT)v;
      } else {  // EPI_RES_ROW_TRANS
        v += p0[o] + p1[(long long)b * aux_bstride + (long long)t * aux_stride + o];
        Out[(long long)b * out_bstride + (long long)o * ld_out + t] = (OT)v;
      }
    }
  }
}

// elementwise f32 -> f16
__global__ void cvt_kernel(const float* __restrict__ in, _Float16* __restrict__ out,
                           long long n) {
  long long i = (long long)blockIdx.x * blockDim.x + threadIdx.x;
  if (i < n) out[i] = (_Float16)in[i];
}

// PPM pooling (1,3,5 -> 35 tokens) from f (b,c,h,w) f32 -> (b,35,c) f16
__global__ void ppm_kernel(const float* __restrict__ f, int C, int H,
                           _Float16* __restrict__ out) {
  int tok = blockIdx.x, b = blockIdx.y;
  int s, idx;
  if (tok == 0)      { s = 1; idx = 0; }
  else if (tok < 10) { s = 3; idx = tok - 1; }
  else               { s = 5; idx = tok - 10; }
  int i = idx / s, j = idx % s;
  int hs = (i * H) / s, he = ((i + 1) * H + s - 1) / s;
  int ws = (j * H) / s, we = ((j + 1) * H + s - 1) / s;
  float inv = 1.f / (float)((he - hs) * (we - ws));
  const float* fb = f + (long long)b * C * H * H;
  for (int ch = threadIdx.x; ch < C; ch += blockDim.x) {
    const float* fc = fb + (long long)ch * H * H;
    float sum = 0.f;
    for (int y = hs; y < he; ++y)
      for (int x = ws; x < we; ++x) sum += fc[y * H + x];
    out[((long long)b * 35 + tok) * C + ch] = (_Float16)(sum * inv);
  }
}

// suffix sum over 4 scale tensors (f16 in/out, f32 accumulate)
__global__ void tailsum_kernel(const _Float16* __restrict__ in,
                               _Float16* __restrict__ out, int sz) {
  int idx = blockIdx.x * blockDim.x + threadIdx.x;
  if (idx >= sz) return;
  float s = 0.f;
  for (int j = 3; j >= 0; --j) {
    s += (float)in[(long long)j * sz + idx];
    out[(long long)j * sz + idx] = (_Float16)s;
  }
}

// LayerNorm over last dim C: f32 in, f16 out; one wave32 per token
__global__ __launch_bounds__(256) void layernorm_kernel(
    const float* __restrict__ x, const float* __restrict__ g,
    const float* __restrict__ bt, _Float16* __restrict__ out, long long ntok, int C) {
  int wv = threadIdx.x >> 5, lane = threadIdx.x & 31;
  long long tok = (long long)blockIdx.x * 8 + wv;
  if (tok >= ntok) return;
  const float* xr = x + tok * C;
  float s = 0.f, s2 = 0.f;
  for (int c = lane; c < C; c += 32) { float v = xr[c]; s += v; s2 += v * v; }
  #pragma unroll
  for (int off = 16; off > 0; off >>= 1) {
    s += __shfl_xor(s, off, 32);
    s2 += __shfl_xor(s2, off, 32);
  }
  float mu  = s / (float)C;
  float var = s2 / (float)C - mu * mu;
  float inv = rsqrtf(var + 1e-5f);
  _Float16* orow = out + tok * C;
  for (int c = lane; c < C; c += 32)
    orow[c] = (_Float16)((xr[c] - mu) * inv * g[c] + bt[c]);
}

// ESA attention: q (b,n,512) f16 x kc/vc (b,35,512) f16 -> o (b,n,512) f16
// grid (ceil(n/64), 8 heads, 16 batch); 4 waves x 16 query rows each
__global__ __launch_bounds__(128) void attn_kernel(
    const _Float16* __restrict__ q, const _Float16* __restrict__ kc,
    const _Float16* __restrict__ vc, _Float16* __restrict__ o, int n) {
  const int lane = threadIdx.x & 31;
  const int wv   = threadIdx.x >> 5;  // 0..3
  const int head = blockIdx.y;
  const int b    = blockIdx.z;
  const int t0   = blockIdx.x * 64 + wv * 16;
  const int lm   = lane & 15;
  const int lh   = lane >> 4;

  __shared__ float    s_s[4][16][64];
  __shared__ _Float16 s_p[4][16][64];

  // q A-fragments (two K=32 halves of d=64)
  const _Float16* qb = q + (long long)b * n * 512 + head * 64;
  const int trow = t0 + lm;
  v16h qa0 = {}, qa1 = {};
  if (trow < n) {
    const _Float16* qr = qb + (long long)trow * 512;
    #pragma unroll
    for (int g = 0; g < 8; ++g) {
      int kk = (g < 4) ? (lh * 8 + g * 2) : (16 + lh * 8 + (g - 4) * 2);
      h2 p = *(const h2*)(qr + kk);
      qa0[2 * g] = p.x; qa0[2 * g + 1] = p.y;
      h2 r = *(const h2*)(qr + 32 + kk);
      qa1[2 * g] = r.x; qa1[2 * g + 1] = r.y;
    }
  }

  // S = (q k^T) * d^-0.5 over 3 m-tiles (m padded 35 -> 48)
  const _Float16* kb = kc + (long long)b * 35 * 512 + head * 64;
  for (int mt = 0; mt < 3; ++mt) {
    int m = mt * 16 + lm;
    v16h kb0 = {}, kb1 = {};
    if (m < 35) {
      const _Float16* kr = kb + (long long)m * 512;
      #pragma unroll
      for (int g = 0; g < 8; ++g) {
        int kk = (g < 4) ? (lh * 8 + g * 2) : (16 + lh * 8 + (g - 4) * 2);
        h2 p = *(const h2*)(kr + kk);
        kb0[2 * g] = p.x; kb0[2 * g + 1] = p.y;
        h2 r = *(const h2*)(kr + 32 + kk);
        kb1[2 * g] = r.x; kb1[2 * g + 1] = r.y;
      }
    }
    v8f sacc = {};
    sacc = wmma_f16(qa0, kb0, sacc);
    sacc = wmma_f16(qa1, kb1, sacc);
    #pragma unroll
    for (int g = 0; g < 8; ++g)
      s_s[wv][g + 8 * lh][mt * 16 + lm] = sacc[g] * 0.125f;
  }
  __syncthreads();

  // softmax over m=35 (lanes 0..15, one row each); P padded to 64 cols
  if (lane < 16) {
    float mx = -1e30f;
    for (int m = 0; m < 35; ++m) mx = fmaxf(mx, s_s[wv][lane][m]);
    float sum = 0.f;
    for (int m = 0; m < 35; ++m) sum += __expf(s_s[wv][lane][m] - mx);
    float inv = 1.f / sum;
    for (int m = 0; m < 64; ++m)
      s_p[wv][lane][m] = (m < 35) ? (_Float16)(__expf(s_s[wv][lane][m] - mx) * inv)
                                  : (_Float16)0.f;
  }
  __syncthreads();

  // O = P x V (K = m padded to 64, N = d = 64 in four 16-col tiles)
  v16h pa0, pa1;
  #pragma unroll
  for (int g = 0; g < 8; ++g) {
    int kk = (g < 4) ? (lh * 8 + g * 2) : (16 + lh * 8 + (g - 4) * 2);
    h2 p = *(const h2*)&s_p[wv][lm][kk];
    pa0[2 * g] = p.x; pa0[2 * g + 1] = p.y;
    h2 r = *(const h2*)&s_p[wv][lm][32 + kk];
    pa1[2 * g] = r.x; pa1[2 * g + 1] = r.y;
  }
  const _Float16* vb = vc + (long long)b * 35 * 512 + head * 64;
  _Float16* ob = o + (long long)b * n * 512 + head * 64;
  for (int nd = 0; nd < 4; ++nd) {
    int dcol = nd * 16 + lm;
    v16h vb0 = {}, vb1 = {};
    #pragma unroll
    for (int g = 0; g < 8; ++g) {
      int ma = (g < 4) ? (lh * 8 + g * 2) : (16 + lh * 8 + (g - 4) * 2);
      vb0[2 * g]     = (ma < 35)     ? vb[(long long)ma * 512 + dcol]       : (_Float16)0;
      vb0[2 * g + 1] = (ma + 1 < 35) ? vb[(long long)(ma + 1) * 512 + dcol] : (_Float16)0;
      int mb = 32 + ma;
      vb1[2 * g]     = (mb < 35)     ? vb[(long long)mb * 512 + dcol]       : (_Float16)0;
      vb1[2 * g + 1] = (mb + 1 < 35) ? vb[(long long)(mb + 1) * 512 + dcol] : (_Float16)0;
    }
    v8f oacc = {};
    oacc = wmma_f16(pa0, vb0, oacc);
    oacc = wmma_f16(pa1, vb1, oacc);
    #pragma unroll
    for (int g = 0; g < 8; ++g) {
      int t = t0 + g + 8 * lh;
      if (t < n) ob[(long long)t * 512 + nd * 16 + lm] = (_Float16)oacc[g];
    }
  }
}

static inline unsigned cdiv(long long a, long long b) { return (unsigned)((a + b - 1) / b); }

extern "C" void kernel_launch(void* const* d_in, const int* in_sizes, int n_in,
                              void* d_out, int out_size, void* d_ws, size_t ws_size,
                              hipStream_t stream) {
  (void)in_sizes; (void)n_in; (void)out_size; (void)ws_size;
  const int C[4]  = {64, 128, 320, 512};
  const int HW[4] = {56, 28, 14, 7};
  const int NT[4] = {3136, 784, 196, 49};
  const int B     = 16;
  const long long OUTOFF[4] = {0, 3211264, 4816896, 5820416};

  const float *f[4], *qkvw[4], *outw[4], *outb[4], *lng[4], *lnb[4];
  const float *fw1[4], *fb1[4], *fw2[4], *fb2[4];
  for (int i = 0; i < 4; ++i) {
    f[i]    = (const float*)d_in[10 * i + 0];
    qkvw[i] = (const float*)d_in[10 * i + 1];
    outw[i] = (const float*)d_in[10 * i + 2];
    outb[i] = (const float*)d_in[10 * i + 3];
    lng[i]  = (const float*)d_in[10 * i + 4];
    lnb[i]  = (const float*)d_in[10 * i + 5];
    fw1[i]  = (const float*)d_in[10 * i + 6];
    fb1[i]  = (const float*)d_in[10 * i + 7];
    fw2[i]  = (const float*)d_in[10 * i + 8];
    fb2[i]  = (const float*)d_in[10 * i + 9];
  }
  const float* kcat_w  = (const float*)d_in[40];
  const float* kcat_b  = (const float*)d_in[41];
  const float* kcat_g  = (const float*)d_in[42];
  const float* kcat_be = (const float*)d_in[43];
  const float* kcat_m  = (const float*)d_in[44];
  const float* kcat_v  = (const float*)d_in[45];
  const float* vcat_w  = (const float*)d_in[46];
  const float* vcat_b  = (const float*)d_in[47];
  const float* vcat_g  = (const float*)d_in[48];
  const float* vcat_be = (const float*)d_in[49];
  const float* vcat_m  = (const float*)d_in[50];
  const float* vcat_v  = (const float*)d_in[51];

  // ---- workspace carve (16B-aligned blocks) ----
  char* wsb = (char*)d_ws;
  auto a16 = [&](size_t elems) {
    _Float16* p = (_Float16*)wsb; wsb += ((elems * 2 + 15) & ~(size_t)15); return p;
  };
  auto a32 = [&](size_t elems) {
    float* p = (float*)wsb; wsb += ((elems * 4 + 15) & ~(size_t)15); return p;
  };

  _Float16 *qkvw16[4], *outw16[4], *fw116[4], *fw216[4], *f16f[4], *fpool16[4];
  for (int i = 0; i < 4; ++i) qkvw16[i] = a16((size_t)1536 * C[i]);
  for (int i = 0; i < 4; ++i) outw16[i] = a16((size_t)C[i] * 512);
  for (int i = 0; i < 4; ++i) fw116[i]  = a16((size_t)512 * C[i]);
  for (int i = 0; i < 4; ++i) fw216[i]  = a16((size_t)C[i] * 512);
  _Float16* kcatw16 = a16((size_t)4 * 512 * 512);
  _Float16* vcatw16 = a16((size_t)4 * 512 * 512);
  for (int i = 0; i < 4; ++i) f16f[i]   = a16((size_t)B * C[i] * NT[i]);
  for (int i = 0; i < 4; ++i) fpool16[i] = a16((size_t)B * 35 * C[i]);
  const size_t SZ = (size_t)B * 35 * 512;
  _Float16* kpool = a16(4 * SZ);
  _Float16* vpool = a16(4 * SZ);
  _Float16* ksum  = a16(4 * SZ);
  _Float16* vsum  = a16(4 * SZ);
  _Float16* kcb   = a16(4 * SZ);
  _Float16* vcb   = a16(4 * SZ);
  const size_t QSZ = (size_t)B * 3136 * 512;
  _Float16* qbuf = a16(QSZ);  // reused as MLP hidden
  _Float16* obuf = a16(QSZ);
  const size_t YSZ = (size_t)B * 3136 * 64;
  float*    ybuf = a32(YSZ);
  _Float16* xnbuf = a16(YSZ);

  // ---- Phase 0: one-time f32 -> f16 conversions ----
  auto cvt = [&](const float* src, _Float16* dst, long long n) {
    cvt_kernel<<<cdiv(n, 256), 256, 0, stream>>>(src, dst, n);
  };
  for (int i = 0; i < 4; ++i) {
    cvt(qkvw[i], qkvw16[i], (long long)1536 * C[i]);
    cvt(outw[i], outw16[i], (long long)C[i] * 512);
    cvt(fw1[i],  fw116[i],  (long long)512 * C[i]);
    cvt(fw2[i],  fw216[i],  (long long)C[i] * 512);
    cvt(f[i],    f16f[i],   (long long)B * C[i] * NT[i]);
  }
  cvt(kcat_w, kcatw16, (long long)4 * 512 * 512);
  cvt(vcat_w, vcatw16, (long long)4 * 512 * 512);

  // ---- Phase 1: pool f, project pooled K/V ----
  for (int i = 0; i < 4; ++i)
    ppm_kernel<<<dim3(35, B), 256, 0, stream>>>(f[i], C[i], HW[i], fpool16[i]);

  for (int i = 0; i < 4; ++i) {
    gemm_kernel<EPI_NONE, _Float16><<<dim3(8, 1, B), 256, 0, stream>>>(
        fpool16[i], (long long)35 * C[i], C[i], 1,
        qkvw16[i] + (size_t)512 * C[i], C[i],
        kpool + (size_t)i * SZ, (long long)35 * 512, 512,
        35, 512, C[i], nullptr, nullptr, nullptr, nullptr, nullptr, 0, 0);
    gemm_kernel<EPI_NONE, _Float16><<<dim3(8, 1, B), 256, 0, stream>>>(
        fpool16[i], (long long)35 * C[i], C[i], 1,
        qkvw16[i] + (size_t)1024 * C[i], C[i],
        vpool + (size_t)i * SZ, (long long)35 * 512, 512,
        35, 512, C[i], nullptr, nullptr, nullptr, nullptr, nullptr, 0, 0);
  }

  // ---- Phase 2: tail sums + cat conv/BN/ReLU ----
  tailsum_kernel<<<cdiv((long long)SZ, 256), 256, 0, stream>>>(kpool, ksum, (int)SZ);
  tailsum_kernel<<<cdiv((long long)SZ, 256), 256, 0, stream>>>(vpool, vsum, (int)SZ);

  for (int i = 0; i < 4; ++i) {
    gemm_kernel<EPI_BNRELU, _Float16><<<dim3(8, 1, B), 256, 0, stream>>>(
        ksum + (size_t)i * SZ, (long long)35 * 512, 512, 1,
        kcatw16 + (size_t)i * 512 * 512, 512,
        kcb + (size_t)i * SZ, (long long)35 * 512, 512,
        35, 512, 512,
        kcat_b + i * 512, kcat_g + i * 512, kcat_be + i * 512,
        kcat_m + i * 512, kcat_v + i * 512, 0, 0);
    gemm_kernel<EPI_BNRELU, _Float16><<<dim3(8, 1, B), 256, 0, stream>>>(
        vsum + (size_t)i * SZ, (long long)35 * 512, 512, 1,
        vcatw16 + (size_t)i * 512 * 512, 512,
        vcb + (size_t)i * SZ, (long long)35 * 512, 512,
        35, 512, 512,
        vcat_b + i * 512, vcat_g + i * 512, vcat_be + i * 512,
        vcat_m + i * 512, vcat_v + i * 512, 0, 0);
  }

  // ---- Phase 3: per-scale q-proj, attention, MLP ----
  for (int i = 0; i < 4; ++i) {
    const int n = NT[i], c = C[i];
    const unsigned mg = cdiv(n, 64);

    // q = f^T (n x c) * Wq^T -> (b, n, 512) f16;  X from NCHW: ldx_t=1, ldx_c=n
    gemm_kernel<EPI_NONE, _Float16><<<dim3(8, mg, B), 256, 0, stream>>>(
        f16f[i], (long long)c * n, 1, n,
        qkvw16[i], c,
        qbuf, (long long)n * 512, 512,
        n, 512, c, nullptr, nullptr, nullptr, nullptr, nullptr, 0, 0);

    attn_kernel<<<dim3(cdiv(n, 64), 8, B), 128, 0, stream>>>(
        qbuf, kcb + (size_t)i * SZ, vcb + (size_t)i * SZ, obuf, n);

    // out-proj + bias + residual(f, NCHW f32) -> y f32
    gemm_kernel<EPI_RES_STRIDED, float><<<dim3(c / 64, mg, B), 256, 0, stream>>>(
        obuf, (long long)n * 512, 512, 1,
        outw16[i], 512,
        ybuf, (long long)n * c, c,
        n, c, 512,
        outb[i], f[i], nullptr, nullptr, nullptr,
        (long long)c * n, n);

    long long ntok = (long long)B * n;
    layernorm_kernel<<<cdiv(ntok, 8), 256, 0, stream>>>(
        ybuf, lng[i], lnb[i], xnbuf, ntok, c);

    // FFN1 + GELU -> hidden f16 (reuse qbuf)
    gemm_kernel<EPI_GELU, _Float16><<<dim3(8, mg, B), 256, 0, stream>>>(
        xnbuf, (long long)n * c, c, 1,
        fw116[i], c,
        qbuf, (long long)n * 512, 512,
        n, 512, c,
        fb1[i], nullptr, nullptr, nullptr, nullptr, 0, 0);

    // FFN2 + bias + residual(y f32) + transposed NCHW store -> d_out f32
    gemm_kernel<EPI_RES_ROW_TRANS, float><<<dim3(c / 64, mg, B), 256, 0, stream>>>(
        qbuf, (long long)n * 512, 512, 1,
        fw216[i], 512,
        (float*)d_out + OUTOFF[i], (long long)c * n, n,
        n, c, 512,
        fb2[i], ybuf, nullptr, nullptr, nullptr,
        (long long)n * c, c);
  }
}
